// MetaTransformBlock_69157563401010
// MI455X (gfx1250) — compile-verified, matrix operand order checked by reading, not run on hardware
//
#include <hip/hip_runtime.h>
#include <hip/hip_fp16.h>

// MetaTransformBlock for MI455X (gfx1250, wave32, WMMA).
//
// Reference shapes: B=4096, H=256, K=16, D=512 (= 8 x 64).
// Dominant cost: per-expert [B x D] x [D x D] GEMM (34.4 GFLOP).
// Strategy: f16 WMMA (v_wmma_f32_16x16x32_f16) with fp32 accumulation;
// meta_w2 pre-packed once into per-lane B-fragment layout in d_ws (8 MB,
// L2-resident on a 192 MB L2 part); h1 = tanh(x_ext*w1+b1) produced on the
// fly into LDS in A-fragment layout; LayerNorm/gating/theta/x_prime fused.

typedef __attribute__((ext_vector_type(16))) _Float16 v16h;
typedef __attribute__((ext_vector_type(8)))  float    v8f;

#define K_EXP 16
#define D_DIM 512
#define LN_EPS 1e-5f

// ---------------------------------------------------------------------------
// Fragment layout helper (CDNA5 ISA 7.12.2, 16-bit A-matrix 16x32):
//   lanes 0-15  : row = lane,    halves 0-7 -> K 0-7,   halves 8-15 -> K 16-23
//   lanes 16-31 : row = lane-16, halves 0-7 -> K 8-15,  halves 8-15 -> K 24-31
// Inverse map for element (row, kk in 0..31):
//   laneGroup = (kk>>3)&1 ; h = (kk&7) + ((kk>>4)&1)*8 ; lane = laneGroup*16+row
// The same packing is used for B fragments with row -> column (N).
// ---------------------------------------------------------------------------

// Pack meta_w2 [K,D,D] f32 (row-major d,e) -> f16 B-fragments in d_ws.
// Fragment id = (k*32 + nt)*16 + kt   (nt = e-tile of 16, kt = d-tile of 32),
// each fragment = 32 lanes x 16 halves = 1KB, lane-major.
__global__ __launch_bounds__(256) void pack_w2_kernel(
    const float* __restrict__ w2, _Float16* __restrict__ dst) {
  int idx = blockIdx.x * 256 + threadIdx.x;       // k*D*D + d*D + e
  int e = idx & 511;
  int d = (idx >> 9) & 511;
  int k = idx >> 18;
  float v = w2[idx];
  int kt = d >> 5, kk = d & 31;
  int nt = e >> 4, n = e & 15;
  int laneGroup = (kk >> 3) & 1;
  int h = (kk & 7) + ((kk >> 4) & 1) * 8;
  int lane = laneGroup * 16 + n;
  int frag = (k * 32 + nt) * 16 + kt;
  dst[(size_t)frag * 512 + lane * 16 + h] = (_Float16)v;
}

// ---------------------------------------------------------------------------
// Gating: gates = softmax(tanh(h_prev @ gw1 + gb1) @ gw2 + gb2). One block/row.
// ---------------------------------------------------------------------------
__global__ __launch_bounds__(256) void gating_kernel(
    const float* __restrict__ h_prev, const float* __restrict__ gw1,
    const float* __restrict__ gb1, const float* __restrict__ gw2,
    const float* __restrict__ gb2, float* __restrict__ gates) {
  __shared__ float hrow[256];
  __shared__ float hid[256];
  __shared__ float lg[16];
  const int b = blockIdx.x;
  const int j = threadIdx.x;

  hrow[j] = h_prev[(size_t)b * 256 + j];
  __syncthreads();

  float sum = gb1[j];
#pragma unroll 8
  for (int i = 0; i < 256; ++i)
    sum = fmaf(hrow[i], gw1[(size_t)i * 256 + j], sum);  // coalesced over j
  hid[j] = tanhf(sum);
  __syncthreads();

  if (j < 16) {
    float l = gb2[j];
#pragma unroll 8
    for (int i = 0; i < 256; ++i)
      l = fmaf(hid[i], gw2[i * 16 + j], l);
    lg[j] = l;
  }
  __syncthreads();

  if (j < 16) {
    float m = lg[0];
#pragma unroll
    for (int i = 1; i < 16; ++i) m = fmaxf(m, lg[i]);
    float ssum = 0.f;
#pragma unroll
    for (int i = 0; i < 16; ++i) ssum += expf(lg[i] - m);
    gates[(size_t)b * 16 + j] = expf(lg[j] - m) / ssum;
  }
}

// ---------------------------------------------------------------------------
// Main fused kernel: per 16-row batch tile, loop over 16 experts:
//   produce h1 tile (A-fragments, f16) in LDS -> WMMA GEMM vs packed w2
//   -> +b2, tanh -> LayerNorm(512) -> gate-weighted theta accumulation.
// Epilogue: +theta0, write theta, x_prime = sum_i x_l[b,i]*theta[b,i,:].
// 256 threads = 8 waves; wave w owns output columns [w*64, w*64+64).
// ---------------------------------------------------------------------------
__global__ __launch_bounds__(256) void meta_main_kernel(
    const float* __restrict__ x_l,
    const float* __restrict__ x_ext,
    const float* __restrict__ mw1,
    const float* __restrict__ mb1,
    const float* __restrict__ mb2,
    const float* __restrict__ ln_g,
    const float* __restrict__ ln_b,
    const float* __restrict__ theta0,
    const float* __restrict__ gates,
    const _Float16* __restrict__ w2h,
    float* __restrict__ out_xp,
    float* __restrict__ out_theta) {
  __shared__ __align__(32) _Float16 ldsA[8192];  // 16 K-tiles x 1KB A-fragments
  __shared__ float ldsG[256];                    // gates [m][k]
  __shared__ float ldsXE[256];                   // x_ext [m][k]
  __shared__ float ldsXL[16][8];                 // x_l tile
  __shared__ float ldsRed[8][2][16];             // per-wave row {sum, sumsq}
  __shared__ float ldsStat[2][16];               // per-row {mean, rstd}
  __shared__ float ldsXP[16][64];                // x_prime accumulator

  const int tid  = threadIdx.x;
  const int lane = tid & 31;
  const int wv   = tid >> 5;    // wave 0..7
  const int grp  = lane >> 4;   // half-wave group (row group for C/D layout)
  const int nsub = lane & 15;   // column-within-tile for C/D layout
  const int bbase = blockIdx.x * 16;

  for (int i = tid; i < 16 * 64; i += 256) ((float*)ldsXP)[i] = 0.0f;
  {
    int m = tid >> 4, kk = tid & 15;
    ldsG[tid]  = gates[(size_t)(bbase + m) * 16 + kk];
    ldsXE[tid] = x_ext[(size_t)(bbase + m) * 16 + kk];
  }
  if (tid < 128) {
    int m = tid >> 3, ii = tid & 7;
    ldsXL[m][ii] = x_l[(size_t)(bbase + m) * 8 + ii];
  }

  // Column-dependent constants (column = wv*64 + j*16 + nsub, expert-invariant)
  float lg4[4], lb4[4], t0v[4];
#pragma unroll
  for (int j = 0; j < 4; ++j) {
    int col = wv * 64 + j * 16 + nsub;
    lg4[j] = ln_g[col];
    lb4[j] = ln_b[col];
    t0v[j] = theta0[col];
  }

  float acc[4][8];
#pragma unroll
  for (int j = 0; j < 4; ++j)
#pragma unroll
    for (int r = 0; r < 8; ++r) acc[j][r] = 0.0f;

  __syncthreads();

  for (int k = 0; k < K_EXP; ++k) {
    // ---- Producer: h1 = tanh(x_ext*w1 + b1), written in A-fragment layout.
#pragma unroll 4
    for (int i = 0; i < 32; ++i) {
      int vi = tid + i * 256;              // 16 rows x 512 d
      int d = vi & 511;
      int M = vi >> 9;
      float h1v = tanhf(ldsXE[M * 16 + k] * mw1[k * 512 + d] + mb1[k * 512 + d]);
      int kt = d >> 5, kk = d & 31;
      int lgp = (kk >> 3) & 1;
      int hh = (kk & 7) + ((kk >> 4) & 1) * 8;
      ldsA[kt * 512 + (lgp * 16 + M) * 16 + hh] = (_Float16)h1v;
    }
    __syncthreads();  // P

    // ---- GEMM: 16x512 @ 512x(64 cols for this wave), f16 WMMA, fp32 acc.
    v8f c[4] = {};
#pragma unroll
    for (int kt = 0; kt < 16; ++kt) {
      v16h afrag = *(const v16h*)&ldsA[kt * 512 + lane * 16];
      const _Float16* bp =
          w2h + ((size_t)((k * 32 + wv * 4) * 16 + kt)) * 512 + lane * 16;
#pragma unroll
      for (int j = 0; j < 4; ++j) {
        v16h bfrag = *(const v16h*)(bp + (size_t)j * 16 * 512);
        c[j] = __builtin_amdgcn_wmma_f32_16x16x32_f16(
            false, afrag, false, bfrag, (short)0, c[j], false, false);
      }
    }

    // ---- Epilogue: +b2, tanh. C/D layout: row M = r + 8*grp, col = tile*16+nsub.
    float hv[4][8];
#pragma unroll
    for (int j = 0; j < 4; ++j) {
      int col = wv * 64 + j * 16 + nsub;
      float b2v = mb2[k * 512 + col];
#pragma unroll
      for (int r = 0; r < 8; ++r) hv[j][r] = tanhf(c[j][r] + b2v);
    }

    // ---- LayerNorm over the 512 columns of each row (= reference [i,e] dims).
    float s[8], sq[8];
#pragma unroll
    for (int r = 0; r < 8; ++r) { s[r] = 0.f; sq[r] = 0.f; }
#pragma unroll
    for (int j = 0; j < 4; ++j)
#pragma unroll
      for (int r = 0; r < 8; ++r) { s[r] += hv[j][r]; sq[r] += hv[j][r] * hv[j][r]; }
    for (int sh = 1; sh <= 8; sh <<= 1) {  // butterfly within each 16-lane half
#pragma unroll
      for (int r = 0; r < 8; ++r) {
        s[r]  += __shfl_xor(s[r], sh, 32);
        sq[r] += __shfl_xor(sq[r], sh, 32);
      }
    }
    if (nsub == 0) {
#pragma unroll
      for (int r = 0; r < 8; ++r) {
        int M = grp * 8 + r;
        ldsRed[wv][0][M] = s[r];
        ldsRed[wv][1][M] = sq[r];
      }
    }
    __syncthreads();  // R1
    if (tid < 16) {
      float t = 0.f, tq = 0.f;
#pragma unroll
      for (int ww = 0; ww < 8; ++ww) { t += ldsRed[ww][0][tid]; tq += ldsRed[ww][1][tid]; }
      float mean = t * (1.0f / 512.0f);
      float var  = tq * (1.0f / 512.0f) - mean * mean;
      ldsStat[0][tid] = mean;
      ldsStat[1][tid] = rsqrtf(var + LN_EPS);
    }
    __syncthreads();  // R2

    // ---- Normalize, affine, gate-weight, accumulate theta.
#pragma unroll
    for (int r = 0; r < 8; ++r) {
      int M = grp * 8 + r;
      float mean = ldsStat[0][M];
      float rstd = ldsStat[1][M];
      float gate = ldsG[M * 16 + k];
#pragma unroll
      for (int j = 0; j < 4; ++j) {
        float wn = (hv[j][r] - mean) * rstd * lg4[j] + lb4[j];
        acc[j][r] += gate * wn;
      }
    }
    // Hazard ordering: ldsA reads (this k) precede R1; next producer's ldsA
    // writes follow R2 -> no extra barrier needed.
  }

  // ---- Final: theta = acc + theta0; x_prime[b,e] = sum_i x_l[b,i]*theta[b,i*64+e].
  // Wave wv owns columns wv*64.. -> i == wv exactly.
#pragma unroll
  for (int r = 0; r < 8; ++r) {
    int M = grp * 8 + r;
    float xl = ldsXL[M][wv];
#pragma unroll
    for (int j = 0; j < 4; ++j) {
      int col = wv * 64 + j * 16 + nsub;
      float tv = acc[j][r] + t0v[j];
      out_theta[(size_t)(bbase + M) * 512 + col] = tv;
      atomicAdd(&ldsXP[M][j * 16 + nsub], xl * tv);  // ds_add_f32
    }
  }
  __syncthreads();
  for (int i = tid; i < 16 * 64; i += 256) {
    int m = i >> 6, e = i & 63;
    out_xp[(size_t)(bbase + m) * 64 + e] = ldsXP[m][e];
  }
}

// ---------------------------------------------------------------------------
extern "C" void kernel_launch(void* const* d_in, const int* in_sizes, int n_in,
                              void* d_out, int out_size, void* d_ws, size_t ws_size,
                              hipStream_t stream) {
  const float* h_prev = (const float*)d_in[0];   // [4096,256]
  const float* x_l    = (const float*)d_in[1];   // [4096,8]
  const float* x_ext  = (const float*)d_in[2];   // [4096,16]
  const float* mw1    = (const float*)d_in[3];   // [16,512]
  const float* mb1    = (const float*)d_in[4];   // [16,512]
  const float* mw2    = (const float*)d_in[5];   // [16,512,512]
  const float* mb2    = (const float*)d_in[6];   // [16,512]
  const float* gw1    = (const float*)d_in[7];   // [256,256]
  const float* gb1    = (const float*)d_in[8];   // [256]
  const float* gw2    = (const float*)d_in[9];   // [256,16]
  const float* gb2    = (const float*)d_in[10];  // [16]
  const float* ln_g   = (const float*)d_in[11];  // [8,64] flat = column index
  const float* ln_b   = (const float*)d_in[12];
  const float* theta0 = (const float*)d_in[13];  // [1,8,64] flat = column index
  (void)in_sizes; (void)n_in; (void)out_size; (void)ws_size;

  // Workspace: gates (256 KB) | packed f16 w2 fragments (8 MB)
  float*    gates = (float*)d_ws;
  _Float16* w2h   = (_Float16*)((char*)d_ws + (size_t)4096 * 16 * 4);

  float* out_xp    = (float*)d_out;               // [4096,64]
  float* out_theta = out_xp + (size_t)4096 * 64;  // [4096,512]

  pack_w2_kernel<<<(16 * 512 * 512) / 256, 256, 0, stream>>>(mw2, w2h);
  gating_kernel<<<4096, 256, 0, stream>>>(h_prev, gw1, gb1, gw2, gb2, gates);
  meta_main_kernel<<<4096 / 16, 256, 0, stream>>>(
      x_l, x_ext, mw1, mb1, mb2, ln_g, ln_b, theta0, gates, w2h,
      out_xp, out_theta);
}
